// PointNetSetAbstraction_72756745994761
// MI455X (gfx1250) — matrix operand order repaired and checked by reading, not compiled, not run on hardware
//
#include <hip/hip_runtime.h>
#include <math.h>

// ---------------- problem constants ----------------
#define B_    8
#define N_    4096
#define F_    40
#define CP_   48              // combined channels 43 padded to 48 (pad = 0)
#define S_    1024            // NPOINT
#define K_    32              // NSAMPLE
#define NT_   (B_*S_*K_)      // 262144 grouped points
#define CIN_  64              // padded MLP channel stride (47/64 -> 64)

typedef __attribute__((ext_vector_type(2)))  float    v2f;
typedef __attribute__((ext_vector_type(8)))  float    v8f;
typedef __attribute__((ext_vector_type(8)))  _Float16 v8h;
typedef __attribute__((ext_vector_type(16))) _Float16 v16h;

// d_out layout: out0 new_xyz^T [8][3][1024] | out1 new_points [8][128][1024] | out2 new_bin [8][1024][40]
#define OUT1_OFF (B_*3*S_)              // 24576
#define OUT2_OFF (OUT1_OFF + B_*128*S_) // 1073152

// ---------------- stage 1: build padded combined tensor + squared norms ----------------
__global__ void k_prep(const float* __restrict__ xyz, const float* __restrict__ feat,
                       float* __restrict__ comb, float* __restrict__ pnorm) {
  int i = blockIdx.x * blockDim.x + threadIdx.x;       // b*N + n
  if (i >= B_ * N_) return;
  int b = i / N_, n = i % N_;
  float* row = comb + (size_t)i * CP_;
  float s = 0.f;
  for (int c = 0; c < 3; ++c) { float v = xyz[((size_t)b*3 + c)*N_ + n]; row[c] = v; s += v*v; }
  for (int f = 0; f < F_; ++f) { float v = feat[((size_t)b*N_ + n)*F_ + f]; row[3+f] = v; s += v*v; }
  for (int c = 43; c < CP_; ++c) row[c] = 0.f;
  pnorm[i] = s;
}

// ---------------- stage 2: farthest point sampling (one block per batch) ----------------
__global__ void k_fps(const float* __restrict__ comb, int* __restrict__ fpsIdx) {
  __shared__ float cent[CP_];
  __shared__ float rv[512];
  __shared__ int   ri[512];
  __shared__ int   sFar;
  const int b = blockIdx.x, t = threadIdx.x;
  const float* cb = comb + (size_t)b * N_ * CP_;
  float d[8];
  for (int j = 0; j < 8; ++j) d[j] = 1e10f;
  if (t == 0) sFar = 0;
  __syncthreads();
  for (int s = 0; s < S_; ++s) {
    int far = sFar;
    if (t == 0) fpsIdx[b*S_ + s] = far;
    if (t < CP_) cent[t] = cb[(size_t)far*CP_ + t];
    __syncthreads();
    float bv = -1.f; int bi = 0;
    for (int j = 0; j < 8; ++j) {
      int pt = t + j*512;                       // ascending -> first-max tie break locally
      const float* pr = cb + (size_t)pt * CP_;
      float dd = 0.f;
      for (int c = 0; c < CP_; c += 4) {
        float4 p = *(const float4*)(pr + c);
        float4 q = *(const float4*)(cent + c);
        float a0=p.x-q.x, a1=p.y-q.y, a2=p.z-q.z, a3=p.w-q.w;
        dd += a0*a0 + a1*a1 + a2*a2 + a3*a3;
      }
      float nd = fminf(d[j], dd);
      d[j] = nd;
      if (nd > bv) { bv = nd; bi = pt; }
    }
    rv[t] = bv; ri[t] = bi;
    __syncthreads();
    for (int st = 256; st > 0; st >>= 1) {
      if (t < st) {
        float ov = rv[t+st]; int oi = ri[t+st];
        if (ov > rv[t] || (ov == rv[t] && oi < ri[t])) { rv[t] = ov; ri[t] = oi; }
      }
      __syncthreads();
    }
    if (t == 0) sFar = ri[0];
    __syncthreads();
  }
}

// ---------------- stage 3: gather query rows; emit out0 (new_xyz^T) and out2 (new_bin) ----------------
__global__ void k_gather(const float* __restrict__ comb, const float* __restrict__ pnorm,
                         const int* __restrict__ fpsIdx,
                         float* __restrict__ qcomb, float* __restrict__ qnorm,
                         float* __restrict__ out) {
  int i = blockIdx.x * blockDim.x + threadIdx.x;       // b*S + s
  if (i >= B_ * S_) return;
  int b = i / S_, s = i % S_;
  int n = fpsIdx[i];
  const float* src = comb + ((size_t)b*N_ + n) * CP_;
  float* dst = qcomb + (size_t)i * CP_;
  for (int c = 0; c < CP_; ++c) dst[c] = src[c];
  qnorm[i] = pnorm[(size_t)b*N_ + n];
  for (int c = 0; c < 3; ++c) out[((size_t)b*3 + c)*S_ + s] = src[c];
  float* o2 = out + OUT2_OFF;
  for (int f = 0; f < F_; ++f) o2[(size_t)i*F_ + f] = src[3+f];
}

// ---------------- stage 4: distance matrix via V_WMMA_F32_16X16X4_F32 + running top-32 ----------------
// block = 256 threads (8 waves); each block owns 16 queries, sweeps N in chunks of 128 points.
// B-chunks are loaded into registers first (one load clause), then 12 back-to-back WMMAs.
__global__ void k_knn(const float* __restrict__ comb, const float* __restrict__ pnorm,
                      const float* __restrict__ qcomb, const float* __restrict__ qnorm,
                      int* __restrict__ knn) {
  __shared__ float distC[16][128];
  __shared__ float bestD[16][32];
  __shared__ int   bestI[16][32];
  __shared__ float curMax[16];
  __shared__ int   curPos[16];
  const int tid = threadIdx.x;
  const int w = tid >> 5, lane = tid & 31, hi = lane >> 4, lm = lane & 15;
  const int bq = blockIdx.x;
  const int b  = bq / (S_/16);
  const int q0 = (bq % (S_/16)) * 16;
  const float* cb = comb  + (size_t)b * N_ * CP_;
  const float* pn = pnorm + (size_t)b * N_;
  const float* qc = qcomb + ((size_t)b*S_ + q0) * CP_;
  const float* qn = qnorm + (size_t)b*S_ + q0;

  if (tid < 16) { curMax[tid] = 1e30f; curPos[tid] = 0; }
  for (int j = tid; j < 16*32; j += 256) { bestD[j>>5][j&31] = 1e30f; bestI[j>>5][j&31] = 0; }
  __syncthreads();

  // Preload A (16 queries x 4-K chunks): lane holds query row lm, K pair (2*hi, 2*hi+1)
  v2f a[12];
  for (int ck = 0; ck < 12; ++ck)
    a[ck] = *(const v2f*)(qc + (size_t)lm*CP_ + ck*4 + 2*hi);

  for (int pc = 0; pc < N_; pc += 128) {
    const int p = pc + w*16 + lm;
    const float* pr = cb + (size_t)p * CP_;
    __builtin_prefetch(pr + 128*CP_, 0, 1);
    // batch all B-chunk loads (single clause + single wait), plus the point norm
    v2f bb[12];
    for (int ck = 0; ck < 12; ++ck) bb[ck] = *(const v2f*)(pr + ck*4 + 2*hi);
    const float pnv = pn[p];
    // uninterrupted WMMA accumulation chain
    v8f acc = {};
    for (int ck = 0; ck < 12; ++ck)
      acc = __builtin_amdgcn_wmma_f32_16x16x4_f32(false, a[ck], false, bb[ck],
                                                  (short)0, acc, false, false);
    for (int r = 0; r < 8; ++r) {
      int qrow = r + 8*hi;                      // D layout: row M = r + 8*(lane/16)
      distC[qrow][w*16 + lm] = qn[qrow] + pnv - 2.f * acc[r];
    }
    __syncthreads();
    if (tid < 16) {                              // per-query running top-32 (order-free; K is max-pooled)
      float cm = curMax[tid]; int cp = curPos[tid];
      for (int j = 0; j < 128; ++j) {
        float dv = distC[tid][j];
        if (dv < cm) {
          bestD[tid][cp] = dv; bestI[tid][cp] = pc + j;
          cm = bestD[tid][0]; cp = 0;
          for (int m = 1; m < 32; ++m) { float v = bestD[tid][m]; if (v > cm) { cm = v; cp = m; } }
        }
      }
      curMax[tid] = cm; curPos[tid] = cp;
    }
    __syncthreads();
  }
  for (int j = tid; j < 16*32; j += 256) {
    int q = j >> 5, k = j & 31;
    knn[((size_t)b*S_ + q0 + q)*K_ + k] = bestI[q][k];
  }
}

// ---------------- stage 5: grouping -> f16 feature rows [NT][64] (47 real + zero pad) ----------------
__global__ void k_group(const float* __restrict__ comb, const float* __restrict__ qcomb,
                        const float* __restrict__ points, const int* __restrict__ knn,
                        _Float16* __restrict__ featH) {
  int pt = blockIdx.x * blockDim.x + threadIdx.x;
  if (pt >= NT_) return;
  int bs = pt / K_;                              // b*S + s
  int b = bs / S_;
  int n = knn[pt];
  const float* g = comb  + ((size_t)b*N_ + n) * CP_;   // [xyz(3) | feat(40)]
  const float* q = qcomb + (size_t)bs * CP_;
  _Float16* o = featH + (size_t)pt * CIN_;
  float x0 = g[0]-q[0], x1 = g[1]-q[1], x2 = g[2]-q[2];
  float r = sqrtf(x0*x0 + x1*x1 + x2*x2 + 1e-12f);
  o[0] = (_Float16)x0; o[1] = (_Float16)x1; o[2] = (_Float16)x2; o[3] = (_Float16)r;
  for (int f = 0; f < F_; ++f) o[4+f] = (_Float16)(g[3+f] - q[3+f]);
  for (int c = 0; c < 3; ++c) o[44+c] = (_Float16)points[((size_t)b*3 + c)*N_ + n];
  for (int c = 47; c < CIN_; ++c) o[c] = (_Float16)0.f;
}

// ---------------- weight packing to f16, padded to stride 64 ----------------
__global__ void k_packw(const float* __restrict__ W1, const float* __restrict__ W2,
                        const float* __restrict__ W3,
                        _Float16* __restrict__ W1h, _Float16* __restrict__ W2h,
                        _Float16* __restrict__ W3h) {
  int i = blockIdx.x * blockDim.x + threadIdx.x;
  if (i < 64*64) {
    int o = i / 64, c = i % 64;
    W1h[i] = (_Float16)((c < 47) ? W1[o*47 + c] : 0.f);
    W2h[i] = (_Float16)W2[i];
  }
  if (i < 128*64) W3h[i] = (_Float16)W3[i];
}

// ---------------- MLP pass 1: y = W x + b via V_WMMA_F32_16X16X32_F16 ----------------
// Weights staged to LDS via CDNA5 async global->LDS copy; B operand hoisted out of the
// M-tile loop; deterministic per-block BN partials.
template <int COUT>
__global__ void k_mlp1(const _Float16* __restrict__ in, const _Float16* __restrict__ W,
                       const float* __restrict__ bias, _Float16* __restrict__ raw,
                       float* __restrict__ partial) {
  __shared__ _Float16 wlds[COUT*64];
  __shared__ float lsum[8][COUT];
  __shared__ float lsq [8][COUT];
  const int tid = threadIdx.x, w = tid >> 5, lane = tid & 31, hi = lane >> 4, lm = lane & 15;
  for (int j = tid; j < 8*COUT; j += 256) { (&lsum[0][0])[j] = 0.f; (&lsq[0][0])[j] = 0.f; }

  // ---- async copy of the f16 weight matrix into LDS (tracked by ASYNCcnt) ----
  {
    __attribute__((address_space(3))) _Float16* wl3 =
        (__attribute__((address_space(3))) _Float16*)wlds;
    unsigned lbase = (unsigned)(unsigned long long)wl3;
    for (int j = tid*8; j < COUT*64; j += 256*8) {
      unsigned ldsa = lbase + (unsigned)j*2u;
      const _Float16* gp = W + j;
      asm volatile("global_load_async_to_lds_b128 %0, %1, off"
                   :: "v"(ldsa), "v"(gp) : "memory");
    }
    asm volatile("s_wait_asynccnt 0" ::: "memory");
  }
  __syncthreads();

  const size_t pt = (size_t)blockIdx.x*128 + w*16 + lm;     // B-matrix column = this lane's point
  const _Float16* inrow = in + pt * CIN_;
  __builtin_prefetch(inrow + 128*CIN_, 0, 1);
  // B operand is identical for every output M-tile: load once (2 x 32B)
  v16h Bv0 = *(const v16h*)(inrow + 16*hi);
  v16h Bv1 = *(const v16h*)(inrow + 32 + 16*hi);

  for (int mt = 0; mt < COUT/16; ++mt) {
    const _Float16* wr = wlds + (size_t)(mt*16 + lm)*64;     // A row (from LDS)
    v8h ra0 = *(const v8h*)(wr + 16*hi);
    v8h rb0 = *(const v8h*)(wr + 8 + 16*hi);
    v8h ra1 = *(const v8h*)(wr + 32 + 16*hi);
    v8h rb1 = *(const v8h*)(wr + 40 + 16*hi);
    v16h A0, A1;
    for (int i2 = 0; i2 < 8; ++i2) {
      A0[i2] = ra0[i2]; A0[i2+8] = rb0[i2];
      A1[i2] = ra1[i2]; A1[i2+8] = rb1[i2];
    }
    v8f acc = {};
    acc = __builtin_amdgcn_wmma_f32_16x16x32_f16(false, A0, false, Bv0, (short)0, acc, false, false);
    acc = __builtin_amdgcn_wmma_f32_16x16x32_f16(false, A1, false, Bv1, (short)0, acc, false, false);

    const int cbase = mt*16 + 8*hi;             // D rows for this lane
    v8h o;
    float ps[8], pq[8];
    for (int r = 0; r < 8; ++r) {
      float y = acc[r] + bias[cbase + r];
      o[r] = (_Float16)y;
      ps[r] = y; pq[r] = y*y;
    }
    *(v8h*)(raw + pt*COUT + cbase) = o;         // 8 contiguous channels, 16B store
    for (int r = 0; r < 8; ++r) {               // deterministic 16-lane (point) reduction
      float s = ps[r], q = pq[r];
      for (int off = 1; off < 16; off <<= 1) {
        s += __shfl_xor(s, off, 32);
        q += __shfl_xor(q, off, 32);
      }
      if (lm == 0) { lsum[w][cbase + r] = s; lsq[w][cbase + r] = q; }
    }
  }
  __syncthreads();
  for (int c = tid; c < COUT; c += 256) {       // fixed-order wave combine -> per-block partials
    float s = 0.f, q = 0.f;
    for (int ww = 0; ww < 8; ++ww) { s += lsum[ww][c]; q += lsq[ww][c]; }
    partial[(size_t)blockIdx.x*COUT*2 + c]        = s;
    partial[(size_t)blockIdx.x*COUT*2 + COUT + c] = q;
  }
}

// ---------------- BN stats finalize: a = g*rsqrt(var+eps), sh = bt - mean*a ----------------
template <int COUT>
__global__ void k_stats(const float* __restrict__ partial, const float* __restrict__ g,
                        const float* __restrict__ bt, float* __restrict__ ab, int nblk) {
  int c = threadIdx.x;
  if (c >= COUT) return;
  float s = 0.f, q = 0.f;
  for (int bk = 0; bk < nblk; ++bk) {
    s += partial[(size_t)bk*COUT*2 + c];
    q += partial[(size_t)bk*COUT*2 + COUT + c];
  }
  float mean = s / (float)NT_;
  float var  = q / (float)NT_ - mean*mean;
  float a = g[c] * rsqrtf(var + 1e-5f);
  ab[2*c]   = a;
  ab[2*c+1] = bt[c] - mean*a;
}

// ---------------- BN apply + ReLU, in place, 8 halves per thread ----------------
template <int COUT>
__global__ void k_pass2(_Float16* __restrict__ y, const float* __restrict__ ab) {
  size_t i = ((size_t)blockIdx.x*blockDim.x + threadIdx.x) * 8;
  if (i >= (size_t)NT_ * COUT) return;
  int cb = (int)(i % COUT);                      // COUT multiple of 8 -> 8 contiguous channels
  v8h v = *(const v8h*)(y + i);
  v8h o;
  for (int r = 0; r < 8; ++r) {
    float x = (float)v[r]*ab[2*(cb+r)] + ab[2*(cb+r)+1];
    o[r] = (_Float16)fmaxf(x, 0.f);
  }
  *(v8h*)(y + i) = o;
}

// ---------------- max over K and emit out1 [B][128][S] ----------------
__global__ void k_maxpool(const _Float16* __restrict__ y3, float* __restrict__ out1) {
  int i = blockIdx.x * blockDim.x + threadIdx.x;       // (b*S+s)*128 + c
  if (i >= B_*S_*128) return;
  int c = i % 128;
  int bs = i / 128;
  int b = bs / S_, s = bs % S_;
  const _Float16* p = y3 + (size_t)bs*K_*128 + c;
  float m = -1e30f;
  for (int k = 0; k < K_; ++k) m = fmaxf(m, (float)p[(size_t)k*128]);
  out1[((size_t)b*128 + c)*S_ + s] = m;
}

// ---------------- orchestration ----------------
extern "C" void kernel_launch(void* const* d_in, const int* in_sizes, int n_in,
                              void* d_out, int out_size, void* d_ws, size_t ws_size,
                              hipStream_t stream) {
  (void)in_sizes; (void)n_in; (void)out_size; (void)ws_size;
  const float* xyz  = (const float*)d_in[0];
  const float* feat = (const float*)d_in[1];
  const float* pts  = (const float*)d_in[2];
  const float* W1 = (const float*)d_in[3];  const float* b1 = (const float*)d_in[4];
  const float* g1 = (const float*)d_in[5];  const float* bt1 = (const float*)d_in[6];
  const float* W2 = (const float*)d_in[7];  const float* b2 = (const float*)d_in[8];
  const float* g2 = (const float*)d_in[9];  const float* bt2 = (const float*)d_in[10];
  const float* W3 = (const float*)d_in[11]; const float* b3 = (const float*)d_in[12];
  const float* g3 = (const float*)d_in[13]; const float* bt3 = (const float*)d_in[14];
  float* out = (float*)d_out;

  char* wsb = (char*)d_ws;
  size_t off = 0;
  auto carve = [&](size_t bytes) -> char* {
    char* p = wsb + off;
    off = (off + bytes + 255) & ~(size_t)255;
    return p;
  };
  float*     comb   = (float*)    carve((size_t)B_*N_*CP_*4);      // 6.3 MB
  float*     pnorm  = (float*)    carve((size_t)B_*N_*4);
  int*       fpsIdx = (int*)      carve((size_t)B_*S_*4);
  float*     qcomb  = (float*)    carve((size_t)B_*S_*CP_*4);
  float*     qnorm  = (float*)    carve((size_t)B_*S_*4);
  int*       knn    = (int*)      carve((size_t)B_*S_*K_*4);
  _Float16*  W1h    = (_Float16*) carve((size_t)64*64*2);
  _Float16*  W2h    = (_Float16*) carve((size_t)64*64*2);
  _Float16*  W3h    = (_Float16*) carve((size_t)128*64*2);
  float*     ab     = (float*)    carve((size_t)128*2*4);
  float*     part   = (float*)    carve((size_t)2048*128*2*4);     // 2 MB
  _Float16*  featH  = (_Float16*) carve((size_t)NT_*64*2);         // 32 MB (reused as y2)
  _Float16*  y1H    = (_Float16*) carve((size_t)NT_*64*2);         // 32 MB
  _Float16*  y3H    = (_Float16*) carve((size_t)NT_*128*2);        // 64 MB
  _Float16*  y2H    = featH;                                       // alias: featH dead after L1

  k_prep   <<<(B_*N_)/256, 256, 0, stream>>>(xyz, feat, comb, pnorm);
  k_fps    <<<B_, 512, 0, stream>>>(comb, fpsIdx);
  k_gather <<<(B_*S_)/256, 256, 0, stream>>>(comb, pnorm, fpsIdx, qcomb, qnorm, out);
  k_knn    <<<B_*(S_/16), 256, 0, stream>>>(comb, pnorm, qcomb, qnorm, knn);
  k_group  <<<NT_/256, 256, 0, stream>>>(comb, qcomb, pts, knn, featH);
  k_packw  <<<(128*64)/256, 256, 0, stream>>>(W1, W2, W3, W1h, W2h, W3h);

  const int nblk = NT_ / 128;  // 2048
  // Layer 1: 47(->64 pad) -> 64
  k_mlp1<64>  <<<nblk, 256, 0, stream>>>(featH, W1h, b1, y1H, part);
  k_stats<64> <<<1, 64, 0, stream>>>(part, g1, bt1, ab, nblk);
  k_pass2<64> <<<(NT_*64/8)/256, 256, 0, stream>>>(y1H, ab);
  // Layer 2: 64 -> 64
  k_mlp1<64>  <<<nblk, 256, 0, stream>>>(y1H, W2h, b2, y2H, part);
  k_stats<64> <<<1, 64, 0, stream>>>(part, g2, bt2, ab, nblk);
  k_pass2<64> <<<(NT_*64/8)/256, 256, 0, stream>>>(y2H, ab);
  // Layer 3: 64 -> 128
  k_mlp1<128> <<<nblk, 256, 0, stream>>>(y2H, W3h, b3, y3H, part);
  k_stats<128><<<1, 128, 0, stream>>>(part, g3, bt3, ab, nblk);
  k_pass2<128><<<(NT_*128/8)/256, 256, 0, stream>>>(y3H, ab);

  k_maxpool <<<(B_*S_*128)/256, 256, 0, stream>>>(y3H, out + OUT1_OFF);
}
